// MixtureOfExperts_43430709297779
// MI455X (gfx1250) — matrix operand order repaired and checked by reading, not compiled, not run on hardware
//
#include <hip/hip_runtime.h>
#include <hip/hip_bf16.h>

// ---------------------------------------------------------------------------
// MoE (top-2 of 8 experts), D=512, H=2048, sparse-routed, bf16 WMMA compute.
// M-tile = 32 rows/workgroup: each B fragment feeds 2 WMMAs (halves L2 traffic).
// ---------------------------------------------------------------------------

#define NEXP 8
#define TOPK 2
#define DM   512
#define HD   2048
#define MT   32              // token rows per workgroup in expert kernel
#define LDA  (DM + 8)        // padded LDS stride for x tile (bank-spread)
#define LDH  (HD + 8)        // padded LDS stride for h1/h2 tiles

typedef __attribute__((ext_vector_type(16))) __bf16 v16bf;
typedef __attribute__((ext_vector_type(8)))  __bf16 v8bf;
typedef __attribute__((ext_vector_type(8)))  float  v8f;

// ---------------------------------------------------------------------------
// K0: zero per-expert counters
// ---------------------------------------------------------------------------
__global__ void moe_zero_cnt(int* cnt) {
    if (threadIdx.x < NEXP) cnt[threadIdx.x] = 0;
}

// ---------------------------------------------------------------------------
// K1: gating. One wave32 per token: logits = x_row @ gate_w, top-2, softmax,
// scatter (token*2+slot) into per-expert lists.
// ---------------------------------------------------------------------------
__global__ __launch_bounds__(256)
void moe_gate(const float* __restrict__ x, const float* __restrict__ gw,
              int* __restrict__ cnt, unsigned* __restrict__ list,
              float* __restrict__ gatev, int ntok) {
    int gwv  = (blockIdx.x * blockDim.x + threadIdx.x) >> 5;  // wave index = token
    int lane = threadIdx.x & 31;
    if (gwv >= ntok) return;
    const float* xr = x + (size_t)gwv * DM;

    float acc[NEXP];
#pragma unroll
    for (int e = 0; e < NEXP; ++e) acc[e] = 0.f;

#pragma unroll 4
    for (int i = 0; i < DM / 32; ++i) {
        int d = lane + 32 * i;
        float xv = xr[d];
        const float* g = gw + (size_t)d * NEXP;
#pragma unroll
        for (int e = 0; e < NEXP; ++e) acc[e] += xv * g[e];
    }
#pragma unroll
    for (int off = 16; off > 0; off >>= 1)
#pragma unroll
        for (int e = 0; e < NEXP; ++e) acc[e] += __shfl_xor(acc[e], off, 32);

    if (lane == 0) {
        int e0 = 0;
#pragma unroll
        for (int e = 1; e < NEXP; ++e) if (acc[e] > acc[e0]) e0 = e;   // lowest idx on tie
        int e1 = (e0 == 0) ? 1 : 0;
#pragma unroll
        for (int e = 0; e < NEXP; ++e)
            if (e != e0 && acc[e] > acc[e1]) e1 = e;
        float l0 = acc[e0], l1 = acc[e1];
        float b  = __expf(l1 - l0);          // l0 >= l1
        float inv = 1.f / (1.f + b);
        float g0 = inv, g1 = b * inv;

        int p0 = atomicAdd(&cnt[e0], 1);
        list[(size_t)e0 * ntok + p0]  = (unsigned)(gwv * 2 + 0);
        gatev[(size_t)e0 * ntok + p0] = g0;
        int p1 = atomicAdd(&cnt[e1], 1);
        list[(size_t)e1 * ntok + p1]  = (unsigned)(gwv * 2 + 1);
        gatev[(size_t)e1 * ntok + p1] = g1;
    }
}

// ---------------------------------------------------------------------------
// K2: convert fp32 [E][R][C] -> bf16 transposed [E][C][R] (K-contiguous rows
// for WMMA B fragments). 32x32 LDS tile transpose, block (32,8).
// ---------------------------------------------------------------------------
__global__ __launch_bounds__(256)
void moe_convtr(const float* __restrict__ in, __bf16* __restrict__ out,
                int R, int C) {
    __shared__ float tile[32][33];
    int e  = blockIdx.z;
    int c0 = blockIdx.x * 32, r0 = blockIdx.y * 32;
    const float* src = in + (size_t)e * R * C;
    __bf16* dst = out + (size_t)e * C * R;
    int tx = threadIdx.x, ty = threadIdx.y;
#pragma unroll
    for (int i = 0; i < 32; i += 8)
        tile[ty + i][tx] = src[(size_t)(r0 + ty + i) * C + (c0 + tx)];
    __syncthreads();
#pragma unroll
    for (int i = 0; i < 32; i += 8)
        dst[(size_t)(c0 + ty + i) * R + (r0 + tx)] = (__bf16)tile[tx][ty + i];
}

// ---------------------------------------------------------------------------
// WMMA fragment helpers (contiguous k order used by this code, not the ISA table order; wave32).
// A 16x32 bf16: lanes<16 row=lane hold K[k0..k0+15];
//               lanes>=16 row=lane-16 hold K[k0+16..k0+31].
// ---------------------------------------------------------------------------
__device__ __forceinline__ v16bf load_a_frag(const __bf16* A, int lda,
                                             int lane, int k0) {
    int row = lane & 15;
    int hio = (lane >> 4) * 16;
    const __bf16* p = A + row * lda + k0 + hio;
    v8bf lo = *(const v8bf*)p;
    v8bf hi = *(const v8bf*)(p + 8);
    return __builtin_shufflevector(lo, hi, 0,1,2,3,4,5,6,7,8,9,10,11,12,13,14,15);
}
// B 32x16 bf16 from K-contiguous weight rows: lane<16 col=lane K[k0..k0+15];
// lane>=16 col=lane-16 K[k0+16..k0+31]. One 32B vector load per lane.
__device__ __forceinline__ v16bf load_b_frag(const __bf16* wrow, int k0) {
    return *(const v16bf*)(wrow + k0);
}

// ---------------------------------------------------------------------------
// K3: fused expert MLP. Block = 256 threads (8 waves). blockIdx.y = expert,
// blockIdx.x = 32-row tile of the expert's gathered pair list.
// LDS: region0 = xs (phase 1) aliased with h2 (phases 2+); region1 = h1.
// ---------------------------------------------------------------------------
__global__ __launch_bounds__(256)
void moe_expert(const float* __restrict__ x,
                const __bf16* __restrict__ w1t,   // [E][H][D]
                const __bf16* __restrict__ w2t,   // [E][H][H]
                const __bf16* __restrict__ w3t,   // [E][D][H]
                const float* __restrict__ n1,     // [E][H]
                const float* __restrict__ n2,     // [E][H]
                const int* __restrict__ cnt,
                const unsigned* __restrict__ list,
                const float* __restrict__ gatev,
                float* __restrict__ outbuf,       // [ntok*2][D]
                int ntok) {
    const int e     = blockIdx.y;
    const int tilei = blockIdx.x;
    const int count = cnt[e];
    if (tilei * MT >= count) return;

    extern __shared__ char smem[];
    __bf16* h2 = (__bf16*)smem;                       // [MT][LDH]  (phase >= 2)
    __bf16* xs = h2;                                  // [MT][LDA]  (phase 1 only)
    __bf16* h1 = h2 + MT * LDH;                       // [MT][LDH]
    float*    rowss   = (float*)(h1 + MT * LDH);      // [MT]
    float*    rstd    = rowss + MT;                   // [MT]
    float*    gater   = rstd + MT;                    // [MT]
    unsigned* tokslot = (unsigned*)(gater + MT);      // [MT]
    int*      validr  = (int*)(tokslot + MT);         // [MT]

    const int tid  = threadIdx.x;
    const int lane = tid & 31;
    const int wv   = tid >> 5;
    const int hi   = lane >> 4;
    const int cl   = lane & 15;

    if (tid < MT) {
        int p = tilei * MT + tid;
        int valid = p < count;
        int pc = valid ? p : count - 1;
        tokslot[tid] = list[(size_t)e * ntok + pc];
        gater[tid]   = valid ? gatev[(size_t)e * ntok + pc] : 0.f;
        validr[tid]  = valid;
        rowss[tid]   = 0.f;
    }
    __syncthreads();

    // ---- stage x tile (fp32 -> bf16 LDS) -----------------------------------
    for (int idx = tid; idx < MT * DM; idx += 256) {
        int r = idx >> 9, c = idx & (DM - 1);
        unsigned tok = tokslot[r] >> 1;
        xs[r * LDA + c] = (__bf16)x[(size_t)tok * DM + c];
    }
    __syncthreads();

    // ---- GEMM1: a1 = x @ w1[e], gelu, -> h1 --------------------------------
    const __bf16* w1e = w1t + (size_t)e * HD * DM;
    for (int nt = wv; nt < HD / 16; nt += 8) {
        int n0 = nt * 16;
        v8f acc0 = {}, acc1 = {};
        const __bf16* bcol = w1e + (size_t)(n0 + cl) * DM + hi * 16;
        for (int k0 = 0; k0 < DM; k0 += 32) {
            v16bf b  = load_b_frag(bcol, k0);
            v16bf a0 = load_a_frag(xs,            LDA, lane, k0);
            v16bf a1 = load_a_frag(xs + 16 * LDA, LDA, lane, k0);
            acc0 = __builtin_amdgcn_wmma_f32_16x16x32_bf16(false, a0, false, b,
                                                           (short)0, acc0, false, false);
            acc1 = __builtin_amdgcn_wmma_f32_16x16x32_bf16(false, a1, false, b,
                                                           (short)0, acc1, false, false);
        }
        int col = n0 + cl;
#pragma unroll
        for (int j = 0; j < 8; ++j) {
            float v0 = acc0[j];
            float g0 = 0.5f * v0 * (1.f + erff(v0 * 0.70710678118f));
            h1[(j + 8 * hi) * LDH + col] = (__bf16)g0;
            float v1 = acc1[j];
            float g1 = 0.5f * v1 * (1.f + erff(v1 * 0.70710678118f));
            h1[(j + 8 * hi + 16) * LDH + col] = (__bf16)g1;
        }
    }
    __syncthreads();

    // ---- rmsnorm(h1)*n1 ----------------------------------------------------
    {
        int r = tid >> 3, part = tid & 7;     // 8 threads per row, 32 rows
        float s = 0.f;
        for (int c = part * (HD / 8); c < (part + 1) * (HD / 8); ++c) {
            float v = (float)h1[r * LDH + c];
            s += v * v;
        }
        atomicAdd(&rowss[r], s);
    }
    __syncthreads();
    if (tid < MT) rstd[tid] = rsqrtf(rowss[tid] * (1.f / HD) + 1e-6f);
    __syncthreads();
    const float* n1e = n1 + (size_t)e * HD;
    for (int idx = tid; idx < MT * HD; idx += 256) {
        int r = idx >> 11, c = idx & (HD - 1);
        float v = (float)h1[r * LDH + c] * rstd[r] * n1e[c];
        h1[r * LDH + c] = (__bf16)v;
    }
    if (tid < MT) rowss[tid] = 0.f;
    __syncthreads();   // also retires xs before h2 overwrites the region

    // ---- GEMM2: a2 = h1 @ w2[e], gelu, -> h2 -------------------------------
    const __bf16* w2e = w2t + (size_t)e * HD * HD;
    for (int nt = wv; nt < HD / 16; nt += 8) {
        int n0 = nt * 16;
        v8f acc0 = {}, acc1 = {};
        const __bf16* bcol = w2e + (size_t)(n0 + cl) * HD + hi * 16;
        for (int k0 = 0; k0 < HD; k0 += 32) {
            v16bf b  = load_b_frag(bcol, k0);
            v16bf a0 = load_a_frag(h1,            LDH, lane, k0);
            v16bf a1 = load_a_frag(h1 + 16 * LDH, LDH, lane, k0);
            acc0 = __builtin_amdgcn_wmma_f32_16x16x32_bf16(false, a0, false, b,
                                                           (short)0, acc0, false, false);
            acc1 = __builtin_amdgcn_wmma_f32_16x16x32_bf16(false, a1, false, b,
                                                           (short)0, acc1, false, false);
        }
        int col = n0 + cl;
#pragma unroll
        for (int j = 0; j < 8; ++j) {
            float v0 = acc0[j];
            float g0 = 0.5f * v0 * (1.f + erff(v0 * 0.70710678118f));
            h2[(j + 8 * hi) * LDH + col] = (__bf16)g0;
            float v1 = acc1[j];
            float g1 = 0.5f * v1 * (1.f + erff(v1 * 0.70710678118f));
            h2[(j + 8 * hi + 16) * LDH + col] = (__bf16)g1;
        }
    }
    __syncthreads();

    // ---- rmsnorm(h2)*n2, then h1 <- h1 + h2n -------------------------------
    {
        int r = tid >> 3, part = tid & 7;
        float s = 0.f;
        for (int c = part * (HD / 8); c < (part + 1) * (HD / 8); ++c) {
            float v = (float)h2[r * LDH + c];
            s += v * v;
        }
        atomicAdd(&rowss[r], s);
    }
    __syncthreads();
    if (tid < MT) rstd[tid] = rsqrtf(rowss[tid] * (1.f / HD) + 1e-6f);
    __syncthreads();
    const float* n2e = n2 + (size_t)e * HD;
    for (int idx = tid; idx < MT * HD; idx += 256) {
        int r = idx >> 11, c = idx & (HD - 1);
        float h2n = (float)h2[r * LDH + c] * rstd[r] * n2e[c];
        float s = (float)h1[r * LDH + c] + h2n;
        h1[r * LDH + c] = (__bf16)s;   // h1 now holds (h1 + h2) operand
    }
    __syncthreads();

    // ---- GEMM3: out = (h1+h2) @ w3[e], scale by gate, scatter --------------
    const __bf16* w3e = w3t + (size_t)e * DM * HD;
    for (int nt = wv; nt < DM / 16; nt += 8) {
        int n0 = nt * 16;
        v8f acc0 = {}, acc1 = {};
        const __bf16* bcol = w3e + (size_t)(n0 + cl) * HD + hi * 16;
        for (int k0 = 0; k0 < HD; k0 += 32) {
            v16bf b  = load_b_frag(bcol, k0);
            v16bf a0 = load_a_frag(h1,            LDH, lane, k0);
            v16bf a1 = load_a_frag(h1 + 16 * LDH, LDH, lane, k0);
            acc0 = __builtin_amdgcn_wmma_f32_16x16x32_bf16(false, a0, false, b,
                                                           (short)0, acc0, false, false);
            acc1 = __builtin_amdgcn_wmma_f32_16x16x32_bf16(false, a1, false, b,
                                                           (short)0, acc1, false, false);
        }
        int col = n0 + cl;
#pragma unroll
        for (int j = 0; j < 8; ++j) {
            int m0 = j + 8 * hi;
            if (validr[m0]) {
                unsigned id = tokslot[m0];
                outbuf[(size_t)id * DM + col] = gater[m0] * acc0[j];
            }
            int m1 = m0 + 16;
            if (validr[m1]) {
                unsigned id = tokslot[m1];
                outbuf[(size_t)id * DM + col] = gater[m1] * acc1[j];
            }
        }
    }
}

// ---------------------------------------------------------------------------
// K4: out[t][d] = outbuf[t*2+0][d] + outbuf[t*2+1][d]
// ---------------------------------------------------------------------------
__global__ __launch_bounds__(256)
void moe_reduce(const float* __restrict__ outbuf, float* __restrict__ out,
                int total) {
    int i = blockIdx.x * blockDim.x + threadIdx.x;
    if (i >= total) return;
    int t = i >> 9, d = i & (DM - 1);
    out[i] = outbuf[(size_t)(2 * t) * DM + d] + outbuf[(size_t)(2 * t + 1) * DM + d];
}

// ---------------------------------------------------------------------------
extern "C" void kernel_launch(void* const* d_in, const int* in_sizes, int n_in,
                              void* d_out, int out_size, void* d_ws, size_t ws_size,
                              hipStream_t stream) {
    const float* x  = (const float*)d_in[0];   // [N][D]
    const float* gw = (const float*)d_in[1];   // [D][E]
    const float* w1 = (const float*)d_in[2];   // [E][D][H]
    const float* w2 = (const float*)d_in[3];   // [E][H][H]
    const float* w3 = (const float*)d_in[4];   // [E][H][D]
    const float* n1 = (const float*)d_in[5];   // [E][H]
    const float* n2 = (const float*)d_in[6];   // [E][H]
    float* out = (float*)d_out;

    const int ntok = in_sizes[0] / DM;         // 8192

    // workspace layout (256B aligned chunks)
    char* ws = (char*)d_ws;
    size_t off = 0;
    auto take = [&](size_t bytes) { char* p = ws + off; off += (bytes + 255) & ~(size_t)255; return p; };
    int*      cnt    = (int*)take(NEXP * sizeof(int));
    unsigned* list   = (unsigned*)take((size_t)NEXP * ntok * sizeof(unsigned));
    float*    gatev  = (float*)take((size_t)NEXP * ntok * sizeof(float));
    __bf16*   w1t    = (__bf16*)take((size_t)NEXP * HD * DM * sizeof(__bf16));
    __bf16*   w2t    = (__bf16*)take((size_t)NEXP * HD * HD * sizeof(__bf16));
    __bf16*   w3t    = (__bf16*)take((size_t)NEXP * DM * HD * sizeof(__bf16));
    float*    outbuf = (float*)take((size_t)ntok * TOPK * DM * sizeof(float));
    (void)ws_size;

    // K0: reset counters
    moe_zero_cnt<<<1, 32, 0, stream>>>(cnt);

    // K1: gating (one wave per token)
    {
        int blocks = (ntok * 32 + 255) / 256;
        moe_gate<<<blocks, 256, 0, stream>>>(x, gw, cnt, list, gatev, ntok);
    }

    // K2: weight convert + transpose to bf16 K-contiguous layout
    {
        dim3 blk(32, 8);
        moe_convtr<<<dim3(HD / 32, DM / 32, NEXP), blk, 0, stream>>>(w1, w1t, DM, HD); // [E][D][H]->[E][H][D]
        moe_convtr<<<dim3(HD / 32, HD / 32, NEXP), blk, 0, stream>>>(w2, w2t, HD, HD); // [E][H][H]->[E][H][H]
        moe_convtr<<<dim3(DM / 32, HD / 32, NEXP), blk, 0, stream>>>(w3, w3t, HD, DM); // [E][H][D]->[E][D][H]
    }

    // K3: fused expert MLP (h2 region aliases xs; phases separated by barriers)
    {
        size_t smem = (size_t)(2 * MT * LDH) * sizeof(__bf16)
                    + (size_t)MT * 5 * sizeof(float);
        dim3 grid((ntok + MT - 1) / MT, NEXP);
        moe_expert<<<grid, 256, smem, stream>>>(x, w1t, w2t, w3t, n1, n2,
                                                cnt, list, gatev, outbuf, ntok);
    }

    // K4: combine the two expert slots per token
    {
        int total = ntok * DM;
        moe_reduce<<<(total + 255) / 256, 256, 0, stream>>>(outbuf, out, total);
    }
}
